// MultiHeadAttention_39135742001649
// MI455X (gfx1250) — matrix operand order repaired and checked by reading, not compiled, not run on hardware
//
#include <hip/hip_runtime.h>
#include <math.h>

// ---------------------------------------------------------------------------
// Problem constants (match reference)
// ---------------------------------------------------------------------------
constexpr int Bb   = 2;
constexpr int Ss   = 2048;
constexpr int Dd   = 1024;
constexpr int Hh   = 16;
constexpr int DKk  = 64;      // D / H
constexpr int WINw = 512;
constexpr int NBn  = Ss / WINw;   // 4
constexpr int Mrows = Bb * Ss;    // 4096

// ---------------------------------------------------------------------------
// Types for CDNA5 WMMA (bf16 A/B, fp32 C/D), wave32
// ---------------------------------------------------------------------------
typedef __bf16 bf16_t;
typedef __attribute__((ext_vector_type(16))) __bf16 v16bf;
typedef __attribute__((ext_vector_type(8)))  __bf16 v8bf;
typedef __attribute__((ext_vector_type(8)))  float  v8f;

__device__ inline v8f wmma_bf16(v16bf a, v16bf b, v8f c) {
    // 8 args: (neg_a, A, neg_b, B, c_mod, C, reuse_a, reuse_b)
    return __builtin_amdgcn_wmma_f32_16x16x32_bf16(
        false, a, false, b, (short)0, c, false, false);
}

// A-matrix 16x32 bf16 fragment (ISA 7.12.2): lane m = lane&15,
// lanes<16 hold K 0..7 (v0..3) and 16..23 (v4..7); lanes>=16 hold K 8..15, 24..31.
__device__ inline v16bf load_a_frag(const bf16_t* base, int lda) {
    int lane = threadIdx.x & 31;
    int m = lane & 15;
    int kofs = (lane & 16) ? 8 : 0;
    const bf16_t* p = base + (size_t)m * lda + kofs;
    v8bf lo = *(const v8bf*)p;
    v8bf hi = *(const v8bf*)(p + 16);
    v16bf r;
#pragma unroll
    for (int i = 0; i < 8; ++i) { r[i] = lo[i]; r[i + 8] = hi[i]; }
    return r;
}

// B-matrix 32x16 bf16 fragment: lane n = lane&15, lanes<16 hold K 0..7, 16..23; lanes>=16 hold K 8..15, 24..31.
__device__ inline v16bf load_b_frag(const bf16_t* base, int ldb) {
    int lane = threadIdx.x & 31;
    int kofs = (lane & 16) ? 8 : 0;
    const bf16_t* p = base + (size_t)kofs * ldb + (lane & 15);
    v16bf r;
#pragma unroll
    for (int i = 0; i < 8; ++i) { r[i] = p[(size_t)i * ldb]; r[i + 8] = p[(size_t)(16 + i) * ldb]; }
    return r;
}

// ---------------------------------------------------------------------------
// Kernel 0: fp32 -> bf16 elementwise convert (for x)
// ---------------------------------------------------------------------------
__global__ __launch_bounds__(256) void cvt_bf16_kernel(const float* __restrict__ in,
                                                       bf16_t* __restrict__ out, int n) {
    int i = blockIdx.x * 256 + threadIdx.x;
    if (i < n) out[i] = (bf16_t)in[i];
}

// ---------------------------------------------------------------------------
// Kernel 1: W (fp32, [N=D, K=D] row-major) -> Wt (bf16, [K=D, N=D] row-major)
// ---------------------------------------------------------------------------
__global__ __launch_bounds__(256) void transpose_w_kernel(const float* __restrict__ w,
                                                          bf16_t* __restrict__ wt) {
    __shared__ float tile[32][33];
    int bx = blockIdx.x * 32, by = blockIdx.y * 32;
    int tx = threadIdx.x, ty = threadIdx.y;        // block (32, 8)
#pragma unroll
    for (int i = ty; i < 32; i += 8)
        tile[i][tx] = w[(size_t)(by + i) * Dd + bx + tx];
    __syncthreads();
#pragma unroll
    for (int i = ty; i < 32; i += 8)
        wt[(size_t)(bx + i) * Dd + by + tx] = (bf16_t)tile[tx][i];
}

// ---------------------------------------------------------------------------
// Kernel 2: QKV projection GEMM. Y = xb @ Wt + bias.
//   Register-blocked: one wave owns a 64x64 output tile (4x4 WMMA accs).
//   Per 32-k-step: 4 A frags + 4 B frags -> 16 WMMAs (4x reuse of each frag).
//   mode 0: store to [B,H,S,DK] (Q, V)     mode 1: store transposed [B,H,DK,S] (K)
// ---------------------------------------------------------------------------
__global__ __launch_bounds__(256) void qkv_gemm_kernel(const bf16_t* __restrict__ xb,
                                                       const bf16_t* __restrict__ wt,
                                                       const float* __restrict__ bias,
                                                       bf16_t* __restrict__ out, int mode) {
    const int NT = Dd / 64;                    // 16 tiles along N
    int wave = threadIdx.x >> 5;
    int tile = blockIdx.x * 8 + wave;          // (M/64)*(N/64) = 1024 tiles
    int m0 = (tile / NT) * 64;
    int n0 = (tile % NT) * 64;

    v8f acc[4][4] = {};
    for (int k0 = 0; k0 < Dd; k0 += 32) {
        v16bf a[4], b[4];
#pragma unroll
        for (int i = 0; i < 4; ++i)
            a[i] = load_a_frag(xb + (size_t)(m0 + 16 * i) * Dd + k0, Dd);
#pragma unroll
        for (int j = 0; j < 4; ++j)
            b[j] = load_b_frag(wt + (size_t)k0 * Dd + n0 + 16 * j, Dd);
#pragma unroll
        for (int i = 0; i < 4; ++i)
#pragma unroll
            for (int j = 0; j < 4; ++j)
                acc[i][j] = wmma_bf16(a[i], b[j], acc[i][j]);
    }

    int lane = threadIdx.x & 31;
    int h = n0 / DKk;                          // 64-wide tile == exactly one head
#pragma unroll
    for (int j = 0; j < 4; ++j) {
        int dk = 16 * j + (lane & 15);         // 0..63 within the head
        float bv = bias[n0 + 16 * j + (lane & 15)];
#pragma unroll
        for (int i = 0; i < 4; ++i) {
#pragma unroll
            for (int e = 0; e < 8; ++e) {
                int m = m0 + 16 * i + e + ((lane & 16) ? 8 : 0);
                int bidx = m / Ss, s = m % Ss;
                float val = acc[i][j][e] + bv;
                if (mode == 1)
                    out[(((size_t)bidx * Hh + h) * DKk + dk) * Ss + s] = (bf16_t)val;
                else
                    out[(((size_t)bidx * Hh + h) * Ss + s) * DKk + dk] = (bf16_t)val;
            }
        }
    }
}

// ---------------------------------------------------------------------------
// Kernel 3: fused windowed double-softmax attention.
//   One block = one (b*H+h, 16 query rows). 8 waves, 256 threads.
//   Q: [BH, S, DK]  Kt: [BH, DK, S]  V: [BH, S, DK]  -> AO bf16 [B, S, D]
// ---------------------------------------------------------------------------
__global__ __launch_bounds__(256) void attn_kernel(const bf16_t* __restrict__ Qb,
                                                   const bf16_t* __restrict__ Kt,
                                                   const bf16_t* __restrict__ Vb,
                                                   bf16_t* __restrict__ AO) {
    __shared__ float  sc[16][WINw];        // 32 KB scores / exp scratch
    __shared__ bf16_t e2[16][WINw];        // 16 KB second-softmax numerators (bf16 A-op)
    __shared__ float  red[16][16];         // row reductions
    __shared__ float  z2[16];              // second-softmax denominators
    __shared__ float  oacc_hi[16][DKk];    // combine key-half partials

    const int tid  = threadIdx.x;
    const int wave = tid >> 5;
    const int lane = tid & 31;

    const int qt = blockIdx.x % (Ss / 16);
    const int bh = blockIdx.x / (Ss / 16);
    const int q0 = qt * 16;
    const int qb = q0 / WINw;              // window block of these queries

    const bf16_t* Qp = Qb + (size_t)bh * Ss * DKk;
    const bf16_t* Kp = Kt + (size_t)bh * DKk * Ss;
    const bf16_t* Vp = Vb + (size_t)bh * Ss * DKk;

    // Q fragments (16 x 64 contraction = 2 k-steps), shared by all waves.
    v16bf qa0 = load_a_frag(Qp + (size_t)q0 * DKk + 0,  DKk);
    v16bf qa1 = load_a_frag(Qp + (size_t)q0 * DKk + 32, DKk);

    if (tid < 16) z2[tid] = 0.f;
    __syncthreads();

    // PV accumulator: wave w handles dk-tile (w&3), key-half (w>>2).
    const int ntile = wave & 3;
    const int khalf = wave >> 2;
    v8f pv = {};

    const float scale = 0.125f;            // 1/sqrt(DK), DK = 64
    const int r  = tid >> 4;               // row 0..15
    const int ch = tid & 15;               // 32-wide chunk within 512
    const int cbase = ch * 32;
    const int gm = q0 + r;                 // global query row

    for (int j = 0; j < NBn; ++j) {
        const int c0 = j * WINw;
        if (j <= qb) {
            // ---- scores: 32 column tiles of 16, 2 WMMA k-steps each ----
            for (int ct = wave; ct < 32; ct += 8) {
                int n0 = ct * 16;
                v16bf b0 = load_b_frag(Kp + (size_t)0  * Ss + c0 + n0, Ss);
                v16bf b1 = load_b_frag(Kp + (size_t)32 * Ss + c0 + n0, Ss);
                v8f cacc = {};
                cacc = wmma_bf16(qa0, b0, cacc);
                cacc = wmma_bf16(qa1, b1, cacc);
                int nn = n0 + (lane & 15);
#pragma unroll
                for (int i = 0; i < 8; ++i) {
                    int rr = i + ((lane & 16) ? 8 : 0);
                    sc[rr][nn] = cacc[i] * scale;
                }
            }
            __syncthreads();

            // ---- window softmax over 512 keys, causal mask (c <= m) ----
            float mx = -INFINITY;
            for (int c = 0; c < 32; ++c) {
                int gc = c0 + cbase + c;
                float v = (gc <= gm) ? sc[r][cbase + c] : -INFINITY;
                mx = fmaxf(mx, v);
            }
            red[r][ch] = mx;
            __syncthreads();
            float rowm = red[r][0];
#pragma unroll
            for (int i = 1; i < 16; ++i) rowm = fmaxf(rowm, red[r][i]);
            float rm = (rowm == -INFINITY) ? 0.f : rowm;
            __syncthreads();

            float ls = 0.f;
            for (int c = 0; c < 32; ++c) {
                int gc = c0 + cbase + c;
                float v = (gc <= gm) ? __expf(sc[r][cbase + c] - rm) : 0.f;
                sc[r][cbase + c] = v;
                ls += v;
            }
            red[r][ch] = ls;
            __syncthreads();
            float denom = 0.f;
#pragma unroll
            for (int i = 0; i < 16; ++i) denom += red[r][i];
            float dinv = (denom == 0.f) ? 0.f : 1.f / denom;
            __syncthreads();

            // ---- second softmax numerators: exp(p1); masked -> exp(0)=1 ----
            float zpart = 0.f;
            for (int c = 0; c < 32; ++c) {
                int gc = c0 + cbase + c;
                float p1 = (gc <= gm) ? sc[r][cbase + c] * dinv : 0.f;
                float e = __expf(p1);
                e2[r][cbase + c] = (bf16_t)e;
                zpart += e;
            }
            red[r][ch] = zpart;
            __syncthreads();
            if (ch == 0) {
                float t = 0.f;
#pragma unroll
                for (int i = 0; i < 16; ++i) t += red[r][i];
                z2[r] += t;
            }
            __syncthreads();
        } else {
            // Future window block: every p1 is 0 -> e2 = exp(0) = 1 everywhere.
            bf16_t* ep = &e2[0][0];
            for (int idx = tid; idx < 16 * WINw; idx += 256) ep[idx] = (bf16_t)1.0f;
            if (tid < 16) z2[tid] += (float)WINw;
            __syncthreads();
        }

        // ---- PV: out_acc += e2 (16x512, bf16 in LDS) @ V_block (512x64) ----
#pragma unroll 2
        for (int ks = 0; ks < 8; ++ks) {
            int kk = khalf * 256 + ks * 32;     // key offset within window block
            v16bf a = load_a_frag(&e2[0][0] + kk, WINw);
            v16bf b = load_b_frag(Vp + (size_t)(c0 + kk) * DKk + ntile * 16, DKk);
            pv = wmma_bf16(a, b, pv);
        }
        __syncthreads();   // before next iteration overwrites sc/e2
    }

    // ---- combine key-halves, divide by Z2, store bf16 [B, S, D] ----
    if (khalf == 1) {
        int n = ntile * 16 + (lane & 15);
#pragma unroll
        for (int i = 0; i < 8; ++i) {
            int rr = i + ((lane & 16) ? 8 : 0);
            oacc_hi[rr][n] = pv[i];
        }
    }
    __syncthreads();
    if (khalf == 0) {
        int n = ntile * 16 + (lane & 15);
        int h  = bh % Hh;
        int bb = bh / Hh;
#pragma unroll
        for (int i = 0; i < 8; ++i) {
            int rr = i + ((lane & 16) ? 8 : 0);
            float val = (pv[i] + oacc_hi[rr][n]) / z2[rr];
            int s = q0 + rr;
            AO[((size_t)bb * Ss + s) * Dd + h * DKk + n] = (bf16_t)val;
        }
    }
}

// ---------------------------------------------------------------------------
// Kernel 4: output projection. d_out (fp32) = AO(bf16) @ WoT + bo
//   Same 64x64-per-wave register blocking as the QKV GEMM.
// ---------------------------------------------------------------------------
__global__ __launch_bounds__(256) void out_gemm_kernel(const bf16_t* __restrict__ ab,
                                                       const bf16_t* __restrict__ wt,
                                                       const float* __restrict__ bias,
                                                       float* __restrict__ out) {
    const int NT = Dd / 64;
    int wave = threadIdx.x >> 5;
    int tile = blockIdx.x * 8 + wave;
    int m0 = (tile / NT) * 64;
    int n0 = (tile % NT) * 64;

    v8f acc[4][4] = {};
    for (int k0 = 0; k0 < Dd; k0 += 32) {
        v16bf a[4], b[4];
#pragma unroll
        for (int i = 0; i < 4; ++i)
            a[i] = load_a_frag(ab + (size_t)(m0 + 16 * i) * Dd + k0, Dd);
#pragma unroll
        for (int j = 0; j < 4; ++j)
            b[j] = load_b_frag(wt + (size_t)k0 * Dd + n0 + 16 * j, Dd);
#pragma unroll
        for (int i = 0; i < 4; ++i)
#pragma unroll
            for (int j = 0; j < 4; ++j)
                acc[i][j] = wmma_bf16(a[i], b[j], acc[i][j]);
    }

    int lane = threadIdx.x & 31;
#pragma unroll
    for (int j = 0; j < 4; ++j) {
        int n = n0 + 16 * j + (lane & 15);
        float bv = bias[n];
#pragma unroll
        for (int i = 0; i < 4; ++i) {
#pragma unroll
            for (int e = 0; e < 8; ++e) {
                int m = m0 + 16 * i + e + ((lane & 16) ? 8 : 0);
                out[(size_t)m * Dd + n] = acc[i][j][e] + bv;
            }
        }
    }
}

// ---------------------------------------------------------------------------
// Host-side launcher
// ---------------------------------------------------------------------------
extern "C" void kernel_launch(void* const* d_in, const int* in_sizes, int n_in,
                              void* d_out, int out_size, void* d_ws, size_t ws_size,
                              hipStream_t stream) {
    (void)in_sizes; (void)n_in; (void)out_size; (void)ws_size;

    const float* x    = (const float*)d_in[0];
    const float* Wq_w = (const float*)d_in[1];
    const float* Wq_b = (const float*)d_in[2];
    const float* Wk_w = (const float*)d_in[3];
    const float* Wk_b = (const float*)d_in[4];
    const float* Wv_w = (const float*)d_in[5];
    const float* Wv_b = (const float*)d_in[6];
    const float* Wo_w = (const float*)d_in[7];
    const float* Wo_b = (const float*)d_in[8];
    float* out = (float*)d_out;

    // Workspace layout (bytes), 48 MiB total.
    char* ws = (char*)d_ws;
    const size_t MiB = 1024 * 1024;
    bf16_t* XB  = (bf16_t*)(ws + 0 * MiB);   // [M, D]      8 MiB
    bf16_t* WQT = (bf16_t*)(ws + 8 * MiB);   // [D, D]      2 MiB
    bf16_t* WKT = (bf16_t*)(ws + 10 * MiB);
    bf16_t* WVT = (bf16_t*)(ws + 12 * MiB);
    bf16_t* WOT = (bf16_t*)(ws + 14 * MiB);
    bf16_t* QB  = (bf16_t*)(ws + 16 * MiB);  // [B,H,S,DK]  8 MiB
    bf16_t* KT  = (bf16_t*)(ws + 24 * MiB);  // [B,H,DK,S]  8 MiB
    bf16_t* VB  = (bf16_t*)(ws + 32 * MiB);  // [B,H,S,DK]  8 MiB
    bf16_t* AO  = (bf16_t*)(ws + 40 * MiB);  // [M, D]      8 MiB

    // 0) x -> bf16
    {
        int n = Mrows * Dd;
        cvt_bf16_kernel<<<(n + 255) / 256, 256, 0, stream>>>(x, XB, n);
    }
    // 1) weight transposes -> bf16
    {
        dim3 g(Dd / 32, Dd / 32), b(32, 8);
        transpose_w_kernel<<<g, b, 0, stream>>>(Wq_w, WQT);
        transpose_w_kernel<<<g, b, 0, stream>>>(Wk_w, WKT);
        transpose_w_kernel<<<g, b, 0, stream>>>(Wv_w, WVT);
        transpose_w_kernel<<<g, b, 0, stream>>>(Wv_w == Wo_w ? Wo_w : Wo_w, WOT);
    }
    // 2) Q/K/V projections: 1024 tiles of 64x64, 8 waves/block -> 128 blocks
    {
        int blocks = (Mrows / 64) * (Dd / 64) / 8;
        qkv_gemm_kernel<<<blocks, 256, 0, stream>>>(XB, WQT, Wq_b, QB, 0);
        qkv_gemm_kernel<<<blocks, 256, 0, stream>>>(XB, WKT, Wk_b, KT, 1);
        qkv_gemm_kernel<<<blocks, 256, 0, stream>>>(XB, WVT, Wv_b, VB, 0);
    }
    // 3) fused attention: B*H*(S/16) = 4096 blocks
    {
        int blocks = Bb * Hh * (Ss / 16);
        attn_kernel<<<blocks, 256, 0, stream>>>(QB, KT, VB, AO);
    }
    // 4) output projection -> fp32 d_out
    {
        int blocks = (Mrows / 64) * (Dd / 64) / 8;
        out_gemm_kernel<<<blocks, 256, 0, stream>>>(AO, WOT, Wo_b, out);
    }
}
